// CognitiveAttention_38783554683015
// MI455X (gfx1250) — compile-verified, matrix operand order checked by reading, not run on hardware
//
#include <hip/hip_runtime.h>

// ---------------------------------------------------------------------------
// Types for CDNA5 WMMA (gfx1250, wave32)
// ---------------------------------------------------------------------------
typedef __attribute__((ext_vector_type(16))) __bf16 v16bf;
typedef __attribute__((ext_vector_type(8)))  __bf16 v8bf;
typedef __attribute__((ext_vector_type(4)))  __bf16 v4bf;
typedef __attribute__((ext_vector_type(8)))  float  v8f;

// Dimensions (fixed by the reference)
#define BATCH   2
#define SEQ     2048
#define DMODEL  1536
#define NH      16
#define NKV     4
#define HD      96
#define MROWS   (BATCH*SEQ)          // 4096
#define NEG_BIG (-1.0e9f)

// ---------------------------------------------------------------------------
// CDNA5 async copy: per-lane 16B global -> LDS, tracked by ASYNCcnt
// (GLOBAL_LOAD_ASYNC_TO_LDS_B128). LDS byte offset is the low 32 bits of the
// generic pointer (LDS aperture keeps the offset in addr[31:0]).
// ---------------------------------------------------------------------------
__device__ __forceinline__ void async_ld_b128(void* lds, const void* gaddr) {
  unsigned l = (unsigned)(unsigned long long)lds;
  asm volatile("global_load_async_to_lds_b128 %0, %1, off"
               :: "v"(l), "v"(gaddr) : "memory");
}
template<int N>
__device__ __forceinline__ void wait_async_le() {
  asm volatile("s_wait_asynccnt %0" :: "i"(N) : "memory");
}

// Wave id as a compiler-provably uniform (SGPR) value: keeps wave-level
// branches scalar so EXEC stays all-ones around WMMA (ISA 7.12 requirement)
// without saveexec bookkeeping.
__device__ __forceinline__ int uniform_wave_id() {
  return __builtin_amdgcn_readfirstlane((int)threadIdx.x) >> 5;
}

// ---------------------------------------------------------------------------
// Fragment load: 16-bit A/B fragment for v_wmma_*_16x16x32 (ISA 7.12.2):
// lane&15 = row(m)/col(n); VGPR0..3 hold k = half*8+{0..7}, VGPR4..7 hold
// k = 16+half*8+{0..7} (half = lane>>4). rowbase 16B aligned.
// ---------------------------------------------------------------------------
__device__ __forceinline__ v16bf load_frag(const __bf16* rowbase, int half) {
  const v8bf lo = *(const v8bf*)(rowbase + half * 8);
  const v8bf hi = *(const v8bf*)(rowbase + 16 + half * 8);
  return __builtin_shufflevector(lo, hi, 0,1,2,3,4,5,6,7,8,9,10,11,12,13,14,15);
}

__device__ __forceinline__ v8f wmma_bf16(v16bf a, v16bf b, v8f c) {
  return __builtin_amdgcn_wmma_f32_16x16x32_bf16(false, a, false, b,
                                                 (short)0, c, false, false);
}

// ---------------------------------------------------------------------------
// Kernel 0: fp32 -> bf16 conversion (one-time; halves all downstream traffic).
// ---------------------------------------------------------------------------
__global__ __launch_bounds__(256)
void cvt_bf16(const float* __restrict__ src, __bf16* __restrict__ dst) {
  const size_t i4 = (size_t)blockIdx.x * 256 + threadIdx.x;
  const float4 v = ((const float4*)src)[i4];
  v4bf o;
  o[0] = (__bf16)v.x; o[1] = (__bf16)v.y; o[2] = (__bf16)v.z; o[3] = (__bf16)v.w;
  ((v4bf*)dst)[i4] = o;
}

// ---------------------------------------------------------------------------
// Kernel 1/4: C[M,N] = A[M,K] * W[N,K]^T  (bf16 in, f32 out, bf16 WMMA)
// 256 threads (8 waves), block tile 128x128, wave tile 64x32.
// Double-buffered async global->LDS staging; ONE barrier per k-chunk.
// ---------------------------------------------------------------------------
#define LSTR 40   // padded LDS row stride (halves) for a 32-wide k-chunk

template<int N, int K>
__global__ __launch_bounds__(256)
void gemm_bf16(const __bf16* __restrict__ A, const __bf16* __restrict__ W,
               float* __restrict__ C) {
  __shared__ __align__(16) __bf16 As[2][128 * LSTR];
  __shared__ __align__(16) __bf16 Bs[2][128 * LSTR];

  const int t    = threadIdx.x;
  const int lane = t & 31;
  const int wave = uniform_wave_id();
  const int mwave = (wave >> 2) * 64, nwave = (wave & 3) * 32;
  const int mblk = blockIdx.y * 128, nblk = blockIdx.x * 128;
  const int half = lane >> 4, l16 = lane & 15;

  auto issue_tile = [&](int buf, int k0) {
#pragma unroll
    for (int i = 0; i < 2; i++) {
      const int seg = i * 256 + t;
      const int row = seg >> 2, c8 = (seg & 3) * 8;
      async_ld_b128(&As[buf][row * LSTR + c8], A + (size_t)(mblk + row) * K + k0 + c8);
      async_ld_b128(&Bs[buf][row * LSTR + c8], W + (size_t)(nblk + row) * K + k0 + c8);
    }
  };

  v8f acc[4][2];
#pragma unroll
  for (int i = 0; i < 4; i++)
#pragma unroll
    for (int j = 0; j < 2; j++) acc[i][j] = (v8f){};

  constexpr int NK = K / 32;
  issue_tile(0, 0);

  for (int it = 0; it < NK; ++it) {
    const int cur = it & 1;
    wait_async_le<0>();          // current tile landed (overlapped prev compute)
    __syncthreads();             // all waves' writes visible / prev reads done
    if (it + 1 < NK) issue_tile(cur ^ 1, (it + 1) * 32);

    v16bf af[4], bfr[2];
#pragma unroll
    for (int i = 0; i < 4; i++)
      af[i] = load_frag(&As[cur][(mwave + i * 16 + l16) * LSTR], half);
#pragma unroll
    for (int j = 0; j < 2; j++)
      bfr[j] = load_frag(&Bs[cur][(nwave + j * 16 + l16) * LSTR], half);

#pragma unroll
    for (int i = 0; i < 4; i++)
#pragma unroll
      for (int j = 0; j < 2; j++)
        acc[i][j] = wmma_bf16(af[i], bfr[j], acc[i][j]);
  }

  // C/D layout: lane l, VGPR r -> m = r + 8*half, n = l&15. N is constexpr.
#pragma unroll
  for (int i = 0; i < 4; i++)
#pragma unroll
    for (int j = 0; j < 2; j++) {
      const int m0 = mblk + mwave + i * 16 + 8 * half;
      const int n  = nblk + nwave + j * 16 + l16;
      float* cp = C + (size_t)m0 * N + n;
#pragma unroll
      for (int r = 0; r < 8; r++) cp[(size_t)r * N] = acc[i][j][r];
    }
}

// ---------------------------------------------------------------------------
// Kernel 2: per-head RMS norm + RoPE + layout transposes.
// One wave per (row, unit). units: 0..15 Q heads, 16..19 K heads, 20..23 V.
// ---------------------------------------------------------------------------
__global__ __launch_bounds__(256)
void norm_rope(const float* __restrict__ Qp, const float* __restrict__ Kp,
               const float* __restrict__ Vp, const float* __restrict__ cosb,
               const float* __restrict__ sinb, const float* __restrict__ qw,
               const float* __restrict__ kw, __bf16* __restrict__ Qr,
               float* __restrict__ k_out, __bf16* __restrict__ Kb,
               float* __restrict__ v_out, __bf16* __restrict__ Vtb) {
  const int gw   = blockIdx.x * 8 + uniform_wave_id();
  const int lane = threadIdx.x & 31;
  const int row  = gw / 24;              // b*SEQ + s
  const int unit = gw % 24;
  const int b = row >> 11, s = row & 2047;

  if (unit < 20) {
    const bool isQ = (unit < 16);
    const int  h   = isQ ? unit : (unit - 16);
    const float* src = isQ ? (Qp + (size_t)row * (NH * HD) + h * HD)
                           : (Kp + (size_t)row * (NKV * HD) + h * HD);
    const float* wn  = isQ ? qw : kw;

    float x[3]; float ss = 0.f;
#pragma unroll
    for (int i = 0; i < 3; i++) { x[i] = src[lane + i * 32]; ss += x[i] * x[i]; }
#pragma unroll
    for (int m = 1; m < 32; m <<= 1) ss += __shfl_xor(ss, m, 32);
    const float rn = rsqrtf(ss * (1.0f / 96.0f) + 1e-6f);

    float xn[3];
#pragma unroll
    for (int i = 0; i < 3; i++) xn[i] = x[i] * rn * wn[lane + i * 32];

    // rotate_half partners always live in lane^16 (d and d+-48).
    const float sx0 = __shfl_xor(xn[0], 16, 32);
    const float sx1 = __shfl_xor(xn[1], 16, 32);
    const float sx2 = __shfl_xor(xn[2], 16, 32);
    float rh[3];
    if (lane < 16) { rh[0] = -sx1; rh[1] = -sx2; rh[2] = sx0; }
    else           { rh[0] = -sx2; rh[1] = sx0;  rh[2] = sx1; }

#pragma unroll
    for (int i = 0; i < 3; i++) {
      const int d = lane + i * 32;
      const float c  = cosb[(size_t)row * HD + d];
      const float sn = sinb[(size_t)row * HD + d];
      const float o  = xn[i] * c + rh[i] * sn;
      if (isQ) {
        Qr[(((size_t)(b * NH + h) * SEQ + s) * HD) + d] = (__bf16)o;
      } else {
        k_out[(((size_t)(b * NKV + h) * SEQ + s) * HD) + d] = o;
        Kb[(((size_t)(b * NKV + h) * SEQ + s) * HD) + d] = (__bf16)o;
      }
    }
  } else {
    const int kvh = unit - 20;
#pragma unroll
    for (int i = 0; i < 3; i++) {
      const int d = lane + i * 32;
      const float val = Vp[(size_t)row * (NKV * HD) + kvh * HD + d];
      v_out[(((size_t)(b * NKV + kvh) * SEQ + s) * HD) + d] = val;
      Vtb[(((size_t)(b * NKV + kvh) * HD + d) * SEQ) + s]   = (__bf16)val;
    }
  }
}

// ---------------------------------------------------------------------------
// Kernel 3: causal flash attention (bf16 in/out). 2 waves/block, one 16-row
// query block per wave, 32-key chunks. Scores computed TRANSPOSED (A=K, B=Q)
// so the score C-layout converts to the P A-fragment by in-lane bf16 packing.
// Double-buffered async staging; masked work only in the 1-2 diagonal chunks.
// All wave-level control values are SGPR-uniform -> scalar branches only.
// ---------------------------------------------------------------------------
#define QSTR 104   // padded stride (halves) for 96-wide rows
#define VSTR 40    // padded stride (halves) for 32-wide rows

template<bool MASKED>
__device__ __forceinline__ void attn_chunk(
    const int t0, const int srow, const int half, const int l16,
    const __bf16* __restrict__ ks, const __bf16* __restrict__ vs,
    const v16bf* __restrict__ qf, v8f* __restrict__ oacc,
    float& mrun, float& lrun) {
  // scores^T tiles: D[t][s] = sum_d K[t][d] * Q[s][d]
  v8f st[2];
#pragma unroll
  for (int g = 0; g < 2; ++g) {
    v8f c0 = (v8f){};
#pragma unroll
    for (int c = 0; c < 3; c++) {
      v16bf kf = load_frag(ks + (g * 16 + l16) * QSTR + c * 32, half);
      c0 = wmma_bf16(kf, qf[c], c0);
    }
    st[g] = c0;
  }

  const float scal = 0.1020620726159658f;   // 96^-0.5
  float rmax = -1e30f;
#pragma unroll
  for (int g = 0; g < 2; g++)
#pragma unroll
    for (int r = 0; r < 8; r++) {
      float v = st[g][r] * scal;
      if (MASKED) {
        const int tt = t0 + g * 16 + r + 8 * half;
        v += (tt > srow ? NEG_BIG : 0.0f);
      }
      st[g][r] = v;
      rmax = fmaxf(rmax, v);
    }
  rmax = fmaxf(rmax, __shfl_xor(rmax, 16, 32));
  const float mnew = fmaxf(mrun, rmax);
  const float corr = __expf(mrun - mnew);
  float rsum = 0.f;
#pragma unroll
  for (int g = 0; g < 2; g++)
#pragma unroll
    for (int r = 0; r < 8; r++) {
      const float p = __expf(st[g][r] - mnew);
      st[g][r] = p;
      rsum += p;
    }
  rsum += __shfl_xor(rsum, 16, 32);
  lrun = lrun * corr + rsum;
  mrun = mnew;

  // P A-fragment directly from C-layout (identical in-lane k pattern).
  v16bf pf;
#pragma unroll
  for (int i = 0; i < 8; i++) {
    pf[i]     = (__bf16)st[0][i];
    pf[8 + i] = (__bf16)st[1][i];
  }

  // Redistribute per-s rescale factor to O's row mapping s = r+8*half.
  float fcor[8];
#pragma unroll
  for (int r = 0; r < 8; r++) fcor[r] = __shfl(corr, r + 8 * half, 32);

#pragma unroll
  for (int dt = 0; dt < 6; dt++) {
    v16bf vf = load_frag(vs + (dt * 16 + l16) * VSTR, half);
#pragma unroll
    for (int r = 0; r < 8; r++) oacc[dt][r] *= fcor[r];
    oacc[dt] = wmma_bf16(pf, vf, oacc[dt]);
  }
}

__global__ __launch_bounds__(64)
void attn_kernel(const __bf16* __restrict__ Qr, const __bf16* __restrict__ Ko,
                 const __bf16* __restrict__ Vt, __bf16* __restrict__ O) {
  __shared__ __align__(16) __bf16 Qs[2][16 * QSTR];        //  6656 B
  __shared__ __align__(16) __bf16 Ks[2][2][32 * QSTR];     // 26624 B
  __shared__ __align__(16) __bf16 Vs[2][2][96 * VSTR];     // 30720 B

  const int lane = threadIdx.x & 31;
  const int wave = uniform_wave_id();          // SGPR-uniform wave id
  const int half = lane >> 4, l16 = lane & 15;

  const int bh  = blockIdx.x >> 6;             // 64 blocks per (b,h)
  const int sbb = blockIdx.x & 63;
  const int b = bh >> 4, h = bh & 15, kv = h >> 2;
  const int s0 = (sbb * 2 + wave) * 16;        // uniform

  const __bf16* Qbase = Qr + ((size_t)(b * NH + h) * SEQ + s0) * HD;
  const __bf16* Kbase = Ko + ((size_t)(b * NKV + kv) * SEQ) * HD;
  const __bf16* Vtbse = Vt + ((size_t)(b * NKV + kv) * HD) * SEQ;

  auto issue_chunk = [&](int buf, int t0) {
#pragma unroll
    for (int j = 0; j < 12; j++) {
      const int seg = j * 32 + lane;
      const int r = seg / 12, c8 = (seg % 12) * 8;
      async_ld_b128(Ks[wave][buf] + r * QSTR + c8,
                    Kbase + (size_t)(t0 + r) * HD + c8);
      const int d = seg >> 2, c2 = (seg & 3) * 8;
      async_ld_b128(Vs[wave][buf] + d * VSTR + c2,
                    Vtbse + (size_t)d * SEQ + t0 + c2);
    }
  };

  // Prologue: Q tile (6 async segs/lane) then chunk 0 (24 segs/lane).
#pragma unroll
  for (int j = 0; j < 6; j++) {
    const int seg = j * 32 + lane;
    const int r = seg / 12, c8 = (seg % 12) * 8;
    async_ld_b128(Qs[wave] + r * QSTR + c8, Qbase + r * HD + c8);
  }
  issue_chunk(0, 0);
  wait_async_le<24>();                 // in-order completion: Q (first 6) done

  v16bf qf[3];
#pragma unroll
  for (int c = 0; c < 3; c++)
    qf[c] = load_frag(Qs[wave] + l16 * QSTR + c * 32, half);

  v8f oacc[6];
#pragma unroll
  for (int i = 0; i < 6; i++) oacc[i] = (v8f){};
  float mrun = -1e30f, lrun = 0.f;     // stats for s = s0 + l16 (both halves)

  const int srow  = s0 + l16;
  const int nch   = (s0 + 16 + 31) >> 5;   // causal chunk count   (uniform)
  const int nfull = (s0 + 1) >> 5;         // chunks needing no mask (uniform)

  for (int tb = 0; tb < nch; ++tb) {
    const int cur = tb & 1;
    if (tb + 1 < nch) {
      issue_chunk(cur ^ 1, (tb + 1) * 32);  // stream next under current WMMAs
      wait_async_le<24>();                  // oldest 24 (current chunk) landed
    } else {
      wait_async_le<0>();
    }
    const __bf16* ks = Ks[wave][cur];
    const __bf16* vs = Vs[wave][cur];
    if (tb < nfull)   // uniform -> scalar branch, EXEC untouched
      attn_chunk<false>(tb * 32, srow, half, l16, ks, vs, qf, oacc, mrun, lrun);
    else
      attn_chunk<true >(tb * 32, srow, half, l16, ks, vs, qf, oacc, mrun, lrun);
  }

  // Epilogue: normalize and store bf16 O as [b*SEQ+s][h*96+d] (GEMM-ready).
  float fl[8];
#pragma unroll
  for (int r = 0; r < 8; r++) fl[r] = __shfl(lrun, r + 8 * half, 32);
#pragma unroll
  for (int dt = 0; dt < 6; dt++)
#pragma unroll
    for (int r = 0; r < 8; r++) {
      const int sg  = s0 + r + 8 * half;
      const int col = h * HD + dt * 16 + l16;
      O[(size_t)(b * SEQ + sg) * DMODEL + col] = (__bf16)(oacc[dt][r] / fl[r]);
    }
}

// ---------------------------------------------------------------------------
// Host-side launcher
// ---------------------------------------------------------------------------
extern "C" void kernel_launch(void* const* d_in, const int* in_sizes, int n_in,
                              void* d_out, int out_size, void* d_ws, size_t ws_size,
                              hipStream_t stream) {
  const float* hidden = (const float*)d_in[0];
  const float* cosb   = (const float*)d_in[1];
  const float* sinb   = (const float*)d_in[2];
  // d_in[3] = attention_mask (causal; applied analytically in attn_kernel)
  const float* Wq = (const float*)d_in[4];
  const float* Wk = (const float*)d_in[5];
  const float* Wv = (const float*)d_in[6];
  const float* Wo = (const float*)d_in[7];
  const float* qw = (const float*)d_in[8];
  const float* kw = (const float*)d_in[9];

  const size_t OUT_SZ = (size_t)MROWS * DMODEL;         // 6291456
  const size_t KV_SZ  = (size_t)BATCH * NKV * SEQ * HD; // 1572864
  const size_t WQ_SZ  = (size_t)DMODEL * DMODEL;        // 2359296
  const size_t WK_SZ  = (size_t)(NKV * HD) * DMODEL;    // 589824

  float* out   = (float*)d_out;
  float* k_out = out + OUT_SZ;
  float* v_out = k_out + KV_SZ;

  // Workspace carve-up (bf16 staging + fp32 intermediates), ~94 MB total.
  char* p = (char*)d_ws;
  __bf16* Hb  = (__bf16*)p; p += OUT_SZ * 2;
  __bf16* Wqb = (__bf16*)p; p += WQ_SZ * 2;
  __bf16* Wkb = (__bf16*)p; p += WK_SZ * 2;
  __bf16* Wvb = (__bf16*)p; p += WK_SZ * 2;
  __bf16* Wob = (__bf16*)p; p += WQ_SZ * 2;
  __bf16* Qrb = (__bf16*)p; p += OUT_SZ * 2;
  __bf16* Kb  = (__bf16*)p; p += KV_SZ * 2;
  __bf16* Vtb = (__bf16*)p; p += KV_SZ * 2;
  __bf16* Oa  = (__bf16*)p; p += OUT_SZ * 2;
  float*  Qp  = (float*)p;  p += OUT_SZ * 4;
  float*  Kp  = (float*)p;  p += KV_SZ * 4;
  float*  Vp  = (float*)p;  p += KV_SZ * 4;

  // 0) One-time fp32 -> bf16 conversions (halves all downstream traffic).
  cvt_bf16<<<OUT_SZ / 1024, 256, 0, stream>>>(hidden, Hb);
  cvt_bf16<<<WQ_SZ / 1024, 256, 0, stream>>>(Wq, Wqb);
  cvt_bf16<<<WK_SZ / 1024, 256, 0, stream>>>(Wk, Wkb);
  cvt_bf16<<<WK_SZ / 1024, 256, 0, stream>>>(Wv, Wvb);
  cvt_bf16<<<WQ_SZ / 1024, 256, 0, stream>>>(Wo, Wob);

  // 1) QKV projections (bf16 WMMA GEMMs, double-buffered async staging)
  gemm_bf16<DMODEL, DMODEL><<<dim3(DMODEL / 128, MROWS / 128), 256, 0, stream>>>(Hb, Wqb, Qp);
  gemm_bf16<NKV * HD, DMODEL><<<dim3((NKV * HD) / 128, MROWS / 128), 256, 0, stream>>>(Hb, Wkb, Kp);
  gemm_bf16<NKV * HD, DMODEL><<<dim3((NKV * HD) / 128, MROWS / 128), 256, 0, stream>>>(Hb, Wvb, Vp);

  // 2) RMS norm + RoPE + transposes (fp32 k/v to d_out, bf16 staging for attn)
  norm_rope<<<(MROWS * 24) / 8, 256, 0, stream>>>(Qp, Kp, Vp, cosb, sinb, qw, kw,
                                                  Qrb, k_out, Kb, v_out, Vtb);

  // 3) Causal flash attention (bf16 in/out, double-buffered async staging)
  attn_kernel<<<(BATCH * NH * (SEQ / 16)) / 2, 64, 0, stream>>>(Qrb, Kb, Vtb, Oa);

  // 4) Output projection
  gemm_bf16<DMODEL, DMODEL><<<dim3(DMODEL / 128, MROWS / 128), 256, 0, stream>>>(Oa, Wob, out);
}
